// BertAdapter_SLT_49933289783411
// MI455X (gfx1250) — matrix operand order repaired and checked by reading, not compiled, not run on hardware
//
#include <hip/hip_runtime.h>
#include <hip/hip_bf16.h>

typedef __attribute__((ext_vector_type(16))) _Float16 v16h;
typedef __attribute__((ext_vector_type(8)))  float    v8f;

#define MFRAG_ELEMS (24 * 32 * 16)   // 24 K-chunks, 32 lanes, 16 halves
#define NFRAG_ELEMS (48 * 32 * 16)   // 48 N-chunks, 32 lanes, 16 halves

// ---------------------------------------------------------------------------
// Precompute kernel: collapse the TT chain into M (768x5) and N (5x768) and
// store them as f16 WMMA B-operand fragments in the exact per-lane layout
// (M: lanes 0-15 hold K=0..7,16..23, lanes 16-31 K=8..15,24..31; N: lanes 0-15 hold
// K=0..15, lanes 16-31 K=16..31). Columns/rows beyond rank 5 are zero-padded.
// ---------------------------------------------------------------------------
__global__ void tt_precompute_kernel(const float* __restrict__ c0,
                                     const float* __restrict__ c1,
                                     const float* __restrict__ c2,
                                     const float* __restrict__ c3,
                                     const float* __restrict__ c4,
                                     const float* __restrict__ c5,
                                     _Float16* __restrict__ Mfrag,
                                     _Float16* __restrict__ Nfrag) {
    int t = blockIdx.x * blockDim.x + threadIdx.x;
    if (t < MFRAG_ELEMS) {
        // M[kk][x] = sum_{v,w} core0[i,v] * core1[v,j,w] * core2[w,k,x]
        // kk = ((i*8)+j)*8+k, x = output rank index (0..4)
        int h     = t & 15;
        int lane  = (t >> 4) & 31;
        int chunk = t >> 9;
        int n     = lane & 15;                      // B column (x index)
        int kl    = (h & 7) + ((h >> 3) << 4) + ((lane >> 4) << 3);     // K within 32-chunk
        int kk    = chunk * 32 + kl;                // global K (0..767)
        float val = 0.0f;
        if (n < 5) {
            int i = kk >> 6, j = (kk >> 3) & 7, k = kk & 7;
            #pragma unroll
            for (int w = 0; w < 5; ++w) {
                float a1 = 0.0f;
                #pragma unroll
                for (int v = 0; v < 5; ++v)
                    a1 += c0[i * 5 + v] * c1[(v * 8 + j) * 5 + w];
                val += a1 * c2[(w * 8 + k) * 5 + n];
            }
        }
        Mfrag[t] = (_Float16)val;
    } else if (t < MFRAG_ELEMS + NFRAG_ELEMS) {
        // N[x][n] = sum_{y,z} core3[x,p,y] * core4[y,q,z] * core5[z,r]
        // n = p*96 + q*12 + r
        int u     = t - MFRAG_ELEMS;
        int h     = u & 15;
        int lane  = (u >> 4) & 31;
        int chunk = u >> 9;
        int ncol  = lane & 15;
        int k     = (lane < 16) ? h : (16 + h);     // K row (x index, 0..31)
        int n     = chunk * 16 + ncol;              // output column (0..767)
        float val = 0.0f;
        if (k < 5) {
            int p = n / 96, q = (n / 12) & 7, r = n % 12;
            #pragma unroll
            for (int y = 0; y < 5; ++y) {
                float cc = 0.0f;
                #pragma unroll
                for (int z = 0; z < 5; ++z)
                    cc += c4[(y * 8 + q) * 5 + z] * c5[z * 12 + r];
                val += c3[(k * 8 + p) * 5 + y] * cc;
            }
        }
        Nfrag[u] = (_Float16)val;
    }
}

// ---------------------------------------------------------------------------
// Main kernel: one wave per block, 16 batch rows per wave (256 tiles total).
// GEMM1: z(16x16) = x(16x768) * Mfrag     -> 24 x v_wmma_f32_16x16x32_f16
// GEMM2: o(16x768) = z(16x32,pad) * Nfrag -> 48 x v_wmma_f32_16x16x32_f16
// Epilogue: +bias, exact GELU (erff), residual add, store.
// ---------------------------------------------------------------------------
__global__ void __launch_bounds__(32)
tt_main_kernel(const float* __restrict__ x,
               const float* __restrict__ bias,
               const _Float16* __restrict__ Mfrag,
               const _Float16* __restrict__ Nfrag,
               float* __restrict__ out) {
    __shared__ float zbuf[16][8];

    const int lane     = threadIdx.x & 31;
    const int tile     = blockIdx.x;          // 0..255
    const int row_base = tile << 4;           // 16 rows per tile
    const int hs       = lane >> 4;           // half-wave select
    const int lrow     = lane & 15;           // A-operand row for this lane
    const int col      = lane & 15;           // C/D column for this lane

    const float* xrow = x + (size_t)(row_base + lrow) * 768;

    // ---- GEMM1: accumulate z = x * M over K=768 ----
    v8f acc = {};
    #pragma unroll 4
    for (int c = 0; c < 24; ++c) {
        const int kb = c * 32 + hs * 8;
        // A-layout (16-bit, 16x32): lanes<16 hold K kb..kb+7 then kb+16..kb+23
        const float4 f0 = *(const float4*)(xrow + kb);
        const float4 f1 = *(const float4*)(xrow + kb + 4);
        const float4 f2 = *(const float4*)(xrow + kb + 16);
        const float4 f3 = *(const float4*)(xrow + kb + 20);
        v16h a;
        a[0]  = (_Float16)f0.x; a[1]  = (_Float16)f0.y;
        a[2]  = (_Float16)f0.z; a[3]  = (_Float16)f0.w;
        a[4]  = (_Float16)f1.x; a[5]  = (_Float16)f1.y;
        a[6]  = (_Float16)f1.z; a[7]  = (_Float16)f1.w;
        a[8]  = (_Float16)f2.x; a[9]  = (_Float16)f2.y;
        a[10] = (_Float16)f2.z; a[11] = (_Float16)f2.w;
        a[12] = (_Float16)f3.x; a[13] = (_Float16)f3.y;
        a[14] = (_Float16)f3.z; a[15] = (_Float16)f3.w;

        const v16h b = *(const v16h*)(Mfrag + ((c * 32 + lane) << 4));
        acc = __builtin_amdgcn_wmma_f32_16x16x32_f16(
            false, a, false, b, (short)0, acc, false, false);
    }

    // ---- reshape z (C-layout) -> A-layout via LDS (cols 0..4 valid) ----
    if (col < 5) {
        #pragma unroll
        for (int v = 0; v < 8; ++v)
            zbuf[v + 8 * hs][col] = acc[v];
    }
    __syncthreads();
    asm volatile("s_wait_dscnt 0" ::: "memory");

    v16h za = {};
    if (hs == 0) {
        // lane<16, row=lrow: halves 0..7 are K=0..7 -> only K<5 nonzero.
        za[0] = (_Float16)zbuf[lrow][0];
        za[1] = (_Float16)zbuf[lrow][1];
        za[2] = (_Float16)zbuf[lrow][2];
        za[3] = (_Float16)zbuf[lrow][3];
        za[4] = (_Float16)zbuf[lrow][4];
    }

    // ---- GEMM2 + epilogue over 48 output chunks of 16 columns ----
    const float inv_sqrt2 = 0.70710678118654752f;
    for (int n16 = 0; n16 < 48; ++n16) {
        const v16h bn = *(const v16h*)(Nfrag + ((n16 * 32 + lane) << 4));
        v8f o = {};
        o = __builtin_amdgcn_wmma_f32_16x16x32_f16(
            false, za, false, bn, (short)0, o, false, false);

        const int   ocol = n16 * 16 + col;
        const float bv   = bias[ocol];
        #pragma unroll
        for (int v = 0; v < 8; ++v) {
            const size_t idx = (size_t)(row_base + v + 8 * hs) * 768 + ocol;
            const float val = o[v] + bv;
            const float g   = 0.5f * val * (1.0f + erff(val * inv_sqrt2));
            out[idx] = x[idx] + g;   // x re-read hits L2 (whole tensor fits)
        }
    }
}

extern "C" void kernel_launch(void* const* d_in, const int* in_sizes, int n_in,
                              void* d_out, int out_size, void* d_ws, size_t ws_size,
                              hipStream_t stream) {
    const float* x    = (const float*)d_in[0];
    const float* bias = (const float*)d_in[1];
    const float* c0   = (const float*)d_in[2];
    const float* c1   = (const float*)d_in[3];
    const float* c2   = (const float*)d_in[4];
    const float* c3   = (const float*)d_in[5];
    const float* c4   = (const float*)d_in[6];
    const float* c5   = (const float*)d_in[7];

    _Float16* Mfrag = (_Float16*)d_ws;
    _Float16* Nfrag = Mfrag + MFRAG_ELEMS;

    // 1) collapse TT cores into WMMA-ready fragments (tiny)
    const int total = MFRAG_ELEMS + NFRAG_ELEMS;          // 36864
    tt_precompute_kernel<<<(total + 255) / 256, 256, 0, stream>>>(
        c0, c1, c2, c3, c4, c5, Mfrag, Nfrag);

    // 2) fused rank-5 bottleneck GEMMs + GELU + residual
    //    4096 rows / 16 rows-per-wave = 256 single-wave blocks
    tt_main_kernel<<<256, 32, 0, stream>>>(x, bias, Mfrag, Nfrag,
                                           (float*)d_out);
}
